// Informer_35038343201086
// MI455X (gfx1250) — compile-verified, matrix operand order checked by reading, not run on hardware
//
#include <hip/hip_runtime.h>
#include <hip/hip_bf16.h>

// ---------------------------------------------------------------------------
// Types for CDNA5 WMMA (wave32, 16x16x32 bf16 -> f32)
// ---------------------------------------------------------------------------
typedef __attribute__((ext_vector_type(16))) __bf16 bf16x16;
typedef __attribute__((ext_vector_type(8)))  float  f32x8;
typedef __attribute__((ext_vector_type(4)))  unsigned u32x4;

union FragBF { bf16x16 v; u32x4 q[2]; };

// Model constants
#define BN   8
#define CIN  32
#define LS   512
#define DM   512
#define HH   8
#define DH   64
#define DFF  2048
#define EL   3
#define UU   35
#define NCLS 10
#define MR   (BN*LS)        // 4096 token rows

__device__ __forceinline__ __bf16 f2bf(float f) {
  unsigned u = __float_as_uint(f);
  unsigned r = (u + 0x7FFFu + ((u >> 16) & 1u)) >> 16;   // RNE
  union { unsigned short s; __bf16 b; } cv;
  cv.s = (unsigned short)r;
  return cv.b;
}

__device__ __forceinline__ float gelu_exact(float x) {
  return 0.5f * x * (1.0f + erff(x * 0.70710678118654752f));
}

// deterministic "random" sample index (stands in for jax PRNG)
__device__ __forceinline__ int samp_idx(int lq, int j, int layer) {
  unsigned s = (unsigned)lq * 1664525u + (unsigned)j * 22695477u +
               (unsigned)layer * 2654435761u + 12345u;
  s ^= s >> 13; s *= 0x9E3779B1u; s ^= s >> 16;
  return (int)(s & (LS - 1));
}

// CDNA5 async global->LDS copy (16B per lane), tracked by ASYNCcnt
__device__ __forceinline__ void async_ld_b128(unsigned lds_addr, const void* g) {
  asm volatile("global_load_async_to_lds_b128 %0, %1, off"
               :: "v"(lds_addr), "v"((unsigned long long)(uintptr_t)g)
               : "memory");
}

// ---------------------------------------------------------------------------
// f32 -> bf16 conversion
// ---------------------------------------------------------------------------
__global__ void cvt_bf16_kernel(const float* __restrict__ src,
                                __bf16* __restrict__ dst, int n) {
  int t = blockIdx.x * blockDim.x + threadIdx.x;
  if (t < n) dst[t] = f2bf(src[t]);
}

// ---------------------------------------------------------------------------
// Conv1d (circular, k=3) + sinusoidal positional encoding
// ---------------------------------------------------------------------------
__global__ void conv_pe_kernel(const float* __restrict__ x_enc,
                               const float* __restrict__ conv_w,
                               float* __restrict__ hf,
                               __bf16* __restrict__ hb) {
  int t = blockIdx.x * blockDim.x + threadIdx.x;   // B*L*DM
  int dch = t & (DM - 1);
  int bl  = t >> 9;               // b*L + l
  int l   = bl & (LS - 1);
  int b   = bl >> 9;
  float acc = 0.f;
  #pragma unroll 4
  for (int c = 0; c < CIN; ++c) {
    const float* xr = x_enc + ((size_t)(b * CIN + c)) * LS;
    const float* wr = conv_w + ((size_t)(dch * CIN + c)) * 3;
    int lm1 = (l - 1 + LS) & (LS - 1);
    int lp1 = (l + 1) & (LS - 1);
    acc += xr[lm1] * wr[0] + xr[l] * wr[1] + xr[lp1] * wr[2];
  }
  int i2 = (dch >> 1) << 1;
  float div = expf(-logf(10000.f) * (float)i2 / (float)DM);
  float ang = (float)l * div;
  acc += (dch & 1) ? cosf(ang) : sinf(ang);
  hf[t] = acc;
  hb[t] = f2bf(acc);
}

// ---------------------------------------------------------------------------
// WMMA GEMM: C[M,N] = A[M,K](bf16) @ W[N,K](bf16)^T + bias
//   block: 256 thr (8 waves) -> 256x64 macro-tile (all waves share N strip)
//   wave : 32x64 (2 M-tiles x 4 N-tiles) -> 8 wmma per 32-K step
//   B strip staged in LDS via async global->LDS b128, double-buffered
// ---------------------------------------------------------------------------
__global__ void __launch_bounds__(256)
gemm_wmma_bf16(const __bf16* __restrict__ A,
               const __bf16* __restrict__ W,
               const float*  __restrict__ bias,
               float* __restrict__ Cf,
               __bf16* __restrict__ Cb,
               int M, int N, int K, int fuse_gelu) {
  __shared__ __align__(16) __bf16 shB[2][64 * 32];   // 2 x 4KB

  int tid  = threadIdx.x;
  int wv   = tid >> 5;          // wave 0..7  -> M sub-block
  int lane = tid & 31;
  int lr   = lane & 15;
  int khalf = (lane >> 4) << 3; // lanes 16..31 take K+8 half

  int tilesN = N >> 6;
  int bm = blockIdx.x / tilesN;       // 256-row block
  int tn = blockIdx.x % tilesN;       // 64-col strip

  // A fragment row pointers (2 M-tiles per wave)
  const __bf16* Ap0 = A + ((size_t)(bm * 256 + wv * 32 + lr)) * K + khalf;
  const __bf16* Ap1 = Ap0 + (size_t)16 * K;

  // async staging assignment: thread -> (row, 16B chunk) of W strip
  int arow = tid >> 2;          // 0..63
  int achk = tid & 3;           // 0..3
  const __bf16* Wg = W + ((size_t)(tn * 64 + arow)) * K + achk * 8;
  unsigned ldsA0 = (unsigned)(uintptr_t)&shB[0][arow * 32 + achk * 8];
  unsigned ldsA1 = (unsigned)(uintptr_t)&shB[1][arow * 32 + achk * 8];

  f32x8 acc[2][4] = {};

  // prologue: stage k0 = 0 into buffer 0
  async_ld_b128(ldsA0, Wg);

  int buf = 0;
  for (int k0 = 0; k0 < K; k0 += 32) {
    if (k0 + 32 < K) {
      async_ld_b128(buf ? ldsA0 : ldsA1, Wg + (k0 + 32));
      asm volatile("s_wait_asynccnt 0x1" ::: "memory");
    } else {
      asm volatile("s_wait_asynccnt 0x0" ::: "memory");
    }
    __syncthreads();   // staged strip visible to all waves

    FragBF fa0, fa1;
    fa0.q[0] = *reinterpret_cast<const u32x4*>(Ap0 + k0);
    fa0.q[1] = *reinterpret_cast<const u32x4*>(Ap0 + k0 + 16);
    fa1.q[0] = *reinterpret_cast<const u32x4*>(Ap1 + k0);
    fa1.q[1] = *reinterpret_cast<const u32x4*>(Ap1 + k0 + 16);
    __builtin_prefetch(Ap0 + k0 + 32, 0, 1);
    __builtin_prefetch(Ap1 + k0 + 32, 0, 1);

    #pragma unroll
    for (int ni = 0; ni < 4; ++ni) {
      const __bf16* sp = &shB[buf][(ni * 16 + lr) * 32 + khalf];
      FragBF fb;
      fb.q[0] = *reinterpret_cast<const u32x4*>(sp);        // ds_load_b128
      fb.q[1] = *reinterpret_cast<const u32x4*>(sp + 16);
      acc[0][ni] = __builtin_amdgcn_wmma_f32_16x16x32_bf16(
          false, fa0.v, false, fb.v, (short)0, acc[0][ni], false, false);
      acc[1][ni] = __builtin_amdgcn_wmma_f32_16x16x32_bf16(
          false, fa1.v, false, fb.v, (short)0, acc[1][ni], false, false);
    }

    __syncthreads();   // all reads done before next async overwrites buffer
    buf ^= 1;
  }

  // epilogue
  #pragma unroll
  for (int mi = 0; mi < 2; ++mi) {
    int rowBase = bm * 256 + wv * 32 + mi * 16 + ((lane >> 4) << 3);
    #pragma unroll
    for (int ni = 0; ni < 4; ++ni) {
      int col = tn * 64 + ni * 16 + lr;
      float bv = bias ? bias[col] : 0.f;
      #pragma unroll
      for (int i = 0; i < 8; ++i) {
        float vv = acc[mi][ni][i] + bv;
        if (fuse_gelu) vv = gelu_exact(vv);
        size_t off = (size_t)(rowBase + i) * N + col;
        if (Cf) Cf[off] = vv;
        if (Cb) Cb[off] = f2bf(vv);
      }
    }
  }
}

// ---------------------------------------------------------------------------
// Sparsity measure M = max(qk_sampled) - sum(qk_sampled)/Lk   per (b,h,l)
// ---------------------------------------------------------------------------
__global__ void mscore_kernel(const float* __restrict__ q,
                              const float* __restrict__ k,
                              float* __restrict__ Msc, int layer) {
  int t  = blockIdx.x * blockDim.x + threadIdx.x;  // B*H*L
  int l  = t & (LS - 1);
  int bh = t >> 9;
  int h  = bh & (HH - 1);
  int b  = bh >> 3;
  const float* qr = q + ((size_t)(b * LS + l)) * DM + h * DH;
  float mx = -3.0e38f, sm = 0.f;
  for (int j = 0; j < UU; ++j) {
    int idx = samp_idx(l, j, layer);
    const float* kr = k + ((size_t)(b * LS + idx)) * DM + h * DH;
    float d = 0.f;
    #pragma unroll 8
    for (int x = 0; x < DH; ++x) d += qr[x] * kr[x];
    mx = fmaxf(mx, d);
    sm += d;
  }
  Msc[t] = mx - sm * (1.f / (float)LS);
}

// ---------------------------------------------------------------------------
// Top-UU selection per (b,h): iterative argmax with masking
// ---------------------------------------------------------------------------
__global__ void topk_kernel(const float* __restrict__ Msc,
                            int* __restrict__ topIdx) {
  int bh  = blockIdx.x;
  int tid = threadIdx.x;                 // 256
  __shared__ float sM[LS];
  __shared__ float rv[256];
  __shared__ int   ri[256];
  sM[tid]       = Msc[bh * LS + tid];
  sM[tid + 256] = Msc[bh * LS + tid + 256];
  __syncthreads();
  for (int it = 0; it < UU; ++it) {
    float a0 = sM[tid], a1 = sM[tid + 256];
    float bv; int bi;
    if (a0 >= a1) { bv = a0; bi = tid; } else { bv = a1; bi = tid + 256; }
    rv[tid] = bv; ri[tid] = bi;
    __syncthreads();
    for (int s = 128; s > 0; s >>= 1) {
      if (tid < s) {
        if (rv[tid + s] > rv[tid] ||
            (rv[tid + s] == rv[tid] && ri[tid + s] < ri[tid])) {
          rv[tid] = rv[tid + s]; ri[tid] = ri[tid + s];
        }
      }
      __syncthreads();
    }
    if (tid == 0) { topIdx[bh * UU + it] = ri[0]; sM[ri[0]] = -3.0e38f; }
    __syncthreads();
  }
}

// ---------------------------------------------------------------------------
// V mean per (b,h,j)  -> vmean[B*H*DH]
// ---------------------------------------------------------------------------
__global__ void vmean_kernel(const float* __restrict__ v,
                             float* __restrict__ vmean) {
  int bh = blockIdx.x;   // B*H
  int j  = threadIdx.x;  // 64
  int h  = bh & (HH - 1);
  int b  = bh >> 3;
  float s = 0.f;
  for (int l = 0; l < LS; ++l)
    s += v[((size_t)(b * LS + l)) * DM + h * DH + j];
  vmean[bh * DH + j] = s * (1.f / (float)LS);
}

// fill ctx[b,l,:] with broadcast vmean (then attn kernel overwrites top rows)
__global__ void fillctx_kernel(const float* __restrict__ vmean,
                               float* __restrict__ ctx) {
  int t = blockIdx.x * blockDim.x + threadIdx.x;  // B*L*DM
  int c = t & (DM - 1);
  int b = t >> 18;
  ctx[t] = vmean[b * DM + c];
}

// ---------------------------------------------------------------------------
// Full softmax attention for the UU selected queries of each (b,h)
// ---------------------------------------------------------------------------
__global__ void attn_kernel(const float* __restrict__ q,
                            const float* __restrict__ k,
                            const float* __restrict__ v,
                            const int* __restrict__ topIdx,
                            float* __restrict__ ctx) {
  int gid = blockIdx.x;            // B*H*UU
  int ui  = gid % UU;
  int bh  = gid / UU;
  int h   = bh & (HH - 1);
  int b   = bh >> 3;
  int lq  = topIdx[bh * UU + ui];
  int tid = threadIdx.x;           // 128

  __shared__ float qrow[DH];
  __shared__ float sc[LS];
  __shared__ float red[128];

  size_t hoff = (size_t)h * DH;
  if (tid < DH) qrow[tid] = q[((size_t)(b * LS + lq)) * DM + hoff + tid];
  __syncthreads();

  for (int kk = tid; kk < LS; kk += 128) {
    const float* kr = k + ((size_t)(b * LS + kk)) * DM + hoff;
    float d = 0.f;
    #pragma unroll 8
    for (int x = 0; x < DH; ++x) d += qrow[x] * kr[x];
    sc[kk] = d * 0.125f;
  }
  __syncthreads();

  float mx = -3.0e38f;
  for (int kk = tid; kk < LS; kk += 128) mx = fmaxf(mx, sc[kk]);
  red[tid] = mx; __syncthreads();
  for (int s = 64; s > 0; s >>= 1) {
    if (tid < s) red[tid] = fmaxf(red[tid], red[tid + s]);
    __syncthreads();
  }
  mx = red[0];
  __syncthreads();

  float sum = 0.f;
  for (int kk = tid; kk < LS; kk += 128) {
    float e = expf(sc[kk] - mx);
    sc[kk] = e;
    sum += e;
  }
  red[tid] = sum; __syncthreads();
  for (int s = 64; s > 0; s >>= 1) {
    if (tid < s) red[tid] += red[tid + s];
    __syncthreads();
  }
  float inv = 1.f / red[0];

  if (tid < DH) {
    float acc = 0.f;
    for (int kk = 0; kk < LS; ++kk)
      acc += sc[kk] * v[((size_t)(b * LS + kk)) * DM + hoff + tid];
    ctx[((size_t)(b * LS + lq)) * DM + hoff + tid] = acc * inv;
  }
}

// ---------------------------------------------------------------------------
// LayerNorm over cols=512 per row; optional residual add; f32 + bf16 outputs
// ---------------------------------------------------------------------------
__global__ void layernorm_kernel(const float* __restrict__ x,
                                 const float* __restrict__ res,
                                 const float* __restrict__ g,
                                 const float* __restrict__ bta,
                                 float* __restrict__ outf,
                                 __bf16* __restrict__ outb) {
  int row = blockIdx.x, tid = threadIdx.x;   // 256 threads, 512 cols
  __shared__ float red[256];
  __shared__ float s_mean, s_rstd;
  size_t base = (size_t)row * DM;
  float v0 = x[base + tid];
  float v1 = x[base + tid + 256];
  if (res) { v0 += res[base + tid]; v1 += res[base + tid + 256]; }
  red[tid] = v0 + v1; __syncthreads();
  for (int s = 128; s > 0; s >>= 1) {
    if (tid < s) red[tid] += red[tid + s];
    __syncthreads();
  }
  if (tid == 0) s_mean = red[0] * (1.f / (float)DM);
  __syncthreads();
  float m  = s_mean;
  float d0 = v0 - m, d1 = v1 - m;
  red[tid] = d0 * d0 + d1 * d1; __syncthreads();
  for (int s = 128; s > 0; s >>= 1) {
    if (tid < s) red[tid] += red[tid + s];
    __syncthreads();
  }
  if (tid == 0) s_rstd = rsqrtf(red[0] * (1.f / (float)DM) + 1e-5f);
  __syncthreads();
  float r  = s_rstd;
  float o0 = d0 * r * g[tid]       + bta[tid];
  float o1 = d1 * r * g[tid + 256] + bta[tid + 256];
  if (outf) { outf[base + tid] = o0; outf[base + tid + 256] = o1; }
  if (outb) { outb[base + tid] = f2bf(o0); outb[base + tid + 256] = f2bf(o1); }
}

// ---------------------------------------------------------------------------
// Final: out[b,c] = sum_i gelu(hf[b,i]) * proj_w[c,i] + proj_b[c]
// ---------------------------------------------------------------------------
__global__ void proj_kernel(const float* __restrict__ hf,
                            const float* __restrict__ pw,
                            const float* __restrict__ pb,
                            float* __restrict__ out) {
  int b = blockIdx.x / NCLS;
  int c = blockIdx.x % NCLS;
  const int FLAT = LS * DM;  // 262144
  const float* hr = hf + (size_t)b * FLAT;
  const float* wr = pw + (size_t)c * FLAT;
  float acc = 0.f;
  for (int i = threadIdx.x; i < FLAT; i += 256)
    acc += gelu_exact(hr[i]) * wr[i];
  __shared__ float red[256];
  red[threadIdx.x] = acc; __syncthreads();
  for (int s = 128; s > 0; s >>= 1) {
    if (threadIdx.x < s) red[threadIdx.x] += red[threadIdx.x + s];
    __syncthreads();
  }
  if (threadIdx.x == 0) out[b * NCLS + c] = red[0] + pb[c];
}

// ---------------------------------------------------------------------------
// Host driver
// ---------------------------------------------------------------------------
extern "C" void kernel_launch(void* const* d_in, const int* in_sizes, int n_in,
                              void* d_out, int out_size, void* d_ws, size_t ws_size,
                              hipStream_t stream) {
  (void)in_sizes; (void)n_in; (void)out_size; (void)ws_size;

  const float* x_enc  = (const float*)d_in[0];
  const float* conv_w = (const float*)d_in[1];
  const float* wq     = (const float*)d_in[2];
  const float* bq     = (const float*)d_in[3];
  const float* wk     = (const float*)d_in[4];
  const float* bk     = (const float*)d_in[5];
  const float* wv     = (const float*)d_in[6];
  const float* bv     = (const float*)d_in[7];
  const float* wo     = (const float*)d_in[8];
  const float* bo     = (const float*)d_in[9];
  const float* w1     = (const float*)d_in[10];
  const float* b1     = (const float*)d_in[11];
  const float* w2     = (const float*)d_in[12];
  const float* b2     = (const float*)d_in[13];
  const float* ln1_g  = (const float*)d_in[14];
  const float* ln1_b  = (const float*)d_in[15];
  const float* ln2_g  = (const float*)d_in[16];
  const float* ln2_b  = (const float*)d_in[17];
  const float* lnf_g  = (const float*)d_in[18];
  const float* lnf_b  = (const float*)d_in[19];
  const float* proj_w = (const float*)d_in[20];
  const float* proj_b = (const float*)d_in[21];
  float* out = (float*)d_out;

  // ---- carve workspace ----
  char* ws = (char*)d_ws;
  size_t off = 0;
  auto carve = [&](size_t bytes) -> void* {
    void* p = ws + off;
    off = (off + bytes + 255) & ~(size_t)255;
    return p;
  };
  const size_t NTOK = (size_t)MR * DM;        // 2M elements
  float*  h_f   = (float*) carve(NTOK * 4);
  __bf16* h_b   = (__bf16*)carve(NTOK * 2);
  float*  q_f   = (float*) carve(NTOK * 4);
  float*  k_f   = (float*) carve(NTOK * 4);
  float*  v_f   = (float*) carve(NTOK * 4);
  float*  ctx_f = (float*) carve(NTOK * 4);
  __bf16* ctx_b = (__bf16*)carve(NTOK * 2);
  float*  oy_f  = (float*) carve(NTOK * 4);   // shared by O-proj out and FFN out
  float*  x1_f  = (float*) carve(NTOK * 4);
  __bf16* x1_b  = (__bf16*)carve(NTOK * 2);
  __bf16* mid_b = (__bf16*)carve((size_t)MR * DFF * 2);
  float*  hf_f  = (float*) carve(NTOK * 4);
  float*  Msc   = (float*) carve((size_t)BN * HH * LS * 4);
  int*    topI  = (int*)   carve((size_t)BN * HH * UU * 4);
  float*  vmean = (float*) carve((size_t)BN * HH * DH * 4);
  __bf16* wq_b  = (__bf16*)carve((size_t)EL * DM * DM * 2);
  __bf16* wk_b  = (__bf16*)carve((size_t)EL * DM * DM * 2);
  __bf16* wv_b  = (__bf16*)carve((size_t)EL * DM * DM * 2);
  __bf16* wo_b  = (__bf16*)carve((size_t)EL * DM * DM * 2);
  __bf16* w1_b  = (__bf16*)carve((size_t)EL * DFF * DM * 2);
  __bf16* w2_b  = (__bf16*)carve((size_t)EL * DM * DFF * 2);

  auto cblocks = [](size_t n) { return (int)((n + 255) / 256); };

  // ---- convert weights to bf16 (deterministic each call) ----
  int nW = EL * DM * DM;
  cvt_bf16_kernel<<<cblocks(nW), 256, 0, stream>>>(wq, wq_b, nW);
  cvt_bf16_kernel<<<cblocks(nW), 256, 0, stream>>>(wk, wk_b, nW);
  cvt_bf16_kernel<<<cblocks(nW), 256, 0, stream>>>(wv, wv_b, nW);
  cvt_bf16_kernel<<<cblocks(nW), 256, 0, stream>>>(wo, wo_b, nW);
  int nF = EL * DFF * DM;
  cvt_bf16_kernel<<<cblocks(nF), 256, 0, stream>>>(w1, w1_b, nF);
  cvt_bf16_kernel<<<cblocks(nF), 256, 0, stream>>>(w2, w2_b, nF);

  // ---- tokenizer conv + positional encoding ----
  conv_pe_kernel<<<cblocks(NTOK), 256, 0, stream>>>(x_enc, conv_w, h_f, h_b);

  // GEMM grids: 256x64 macro-tile per block (8 waves of 32x64)
  const int gQKV = (MR / 256) * (DM / 64);    // 16*8  = 128
  const int gFF1 = (MR / 256) * (DFF / 64);   // 16*32 = 512

  for (int l = 0; l < EL; ++l) {
    const __bf16* wq_l = wq_b + (size_t)l * DM * DM;
    const __bf16* wk_l = wk_b + (size_t)l * DM * DM;
    const __bf16* wv_l = wv_b + (size_t)l * DM * DM;
    const __bf16* wo_l = wo_b + (size_t)l * DM * DM;
    const __bf16* w1_l = w1_b + (size_t)l * DFF * DM;
    const __bf16* w2_l = w2_b + (size_t)l * DM * DFF;

    // Q, K, V projections
    gemm_wmma_bf16<<<gQKV, 256, 0, stream>>>(h_b, wq_l, bq + l * DM,
                                             q_f, nullptr, MR, DM, DM, 0);
    gemm_wmma_bf16<<<gQKV, 256, 0, stream>>>(h_b, wk_l, bk + l * DM,
                                             k_f, nullptr, MR, DM, DM, 0);
    gemm_wmma_bf16<<<gQKV, 256, 0, stream>>>(h_b, wv_l, bv + l * DM,
                                             v_f, nullptr, MR, DM, DM, 0);

    // ProbSparse attention
    mscore_kernel<<<BN * HH * LS / 256, 256, 0, stream>>>(q_f, k_f, Msc, l);
    topk_kernel<<<BN * HH, 256, 0, stream>>>(Msc, topI);
    vmean_kernel<<<BN * HH, DH, 0, stream>>>(v_f, vmean);
    fillctx_kernel<<<cblocks(NTOK), 256, 0, stream>>>(vmean, ctx_f);
    attn_kernel<<<BN * HH * UU, 128, 0, stream>>>(q_f, k_f, v_f, topI, ctx_f);
    cvt_bf16_kernel<<<cblocks(NTOK), 256, 0, stream>>>(ctx_f, ctx_b, (int)NTOK);

    // Output projection + LN1 (with residual h)
    gemm_wmma_bf16<<<gQKV, 256, 0, stream>>>(ctx_b, wo_l, bo + l * DM,
                                             oy_f, nullptr, MR, DM, DM, 0);
    layernorm_kernel<<<MR, 256, 0, stream>>>(h_f, oy_f, ln1_g + l * DM,
                                             ln1_b + l * DM, x1_f, x1_b);

    // FFN: gelu(x1 @ w1^T + b1) @ w2^T + b2, then LN2 (residual x1) -> h
    gemm_wmma_bf16<<<gFF1, 256, 0, stream>>>(x1_b, w1_l, b1 + l * DFF,
                                             nullptr, mid_b, MR, DFF, DM, 1);
    gemm_wmma_bf16<<<gQKV, 256, 0, stream>>>(mid_b, w2_l, b2 + l * DM,
                                             oy_f, nullptr, MR, DM, DFF, 0);
    layernorm_kernel<<<MR, 256, 0, stream>>>(x1_f, oy_f, ln2_g + l * DM,
                                             ln2_b + l * DM, h_f, h_b);
  }

  // final LN -> gelu (fused in proj) -> projection
  layernorm_kernel<<<MR, 256, 0, stream>>>(h_f, nullptr, lnf_g, lnf_b,
                                           hf_f, nullptr);
  proj_kernel<<<BN * NCLS, 256, 0, stream>>>(hf_f, proj_w, proj_b, out);
}
